// Model_39676907888096
// MI455X (gfx1250) — matrix operand order repaired and checked by reading, not compiled, not run on hardware
//
#include <hip/hip_runtime.h>
#include <hip/hip_bf16.h>

// CDNA5 / gfx1250: wave32, WMMA bf16 with f32 accumulate.
typedef __attribute__((ext_vector_type(16))) __bf16       v16bf;
typedef __attribute__((ext_vector_type(8)))  float        v8f;
typedef __attribute__((ext_vector_type(4)))  unsigned int u32x4;
typedef __attribute__((ext_vector_type(4)))  float        f32x4;

// Fragment = 16 bf16 = 8 dwords = 2 quad-dwords. All our layouts make the two
// 8-element halves contiguous and 16B-aligned -> build with two b128 loads.
union Frag16 {
    v16bf        v;
    unsigned int u[8];
    u32x4        q[2];
};

union PackBF2 { __bf16 h[2]; unsigned int w; };

static __device__ __forceinline__ unsigned int pack2(float lo, float hi) {
    PackBF2 pk;
    pk.h[0] = (__bf16)lo;
    pk.h[1] = (__bf16)hi;
    return pk.w;
}

static __device__ __forceinline__ v8f wmma_bf16(v16bf a, v16bf b, v8f c) {
    // (neg_a, A, neg_b, B, c_mod, C, reuse_a, reuse_b)
    return __builtin_amdgcn_wmma_f32_16x16x32_bf16(false, a, false, b, (short)0, c, false, false);
}

// ---------------------------------------------------------------------------
// Tiled WMMA GEMM:  C[M,N] = A[M,K] (f32 or bf16) * B_f32[K,N]  (B cast->bf16)
// Block tile 128x128, K-tile 64, 256 threads = 8 waves (4y x 2x); each wave
// computes a 32x64 sub-tile = 2x4 WMMA accumulators.
// Software pipeline: tile t+1 is loaded into REGISTERS while WMMAs consume
// tile t from LDS (global latency hidden behind the XDL pipe).
// LDS rows are 72 bf16 = 144 B = 9*16 B so fragment runs stay 16B aligned.
// If `vt` is non-null (QKV GEMM), output columns >= 2048 (the V part) are
// written TRANSPOSED to vt[b][d][s] so attention can pair-load V over keys.
// ---------------------------------------------------------------------------
template<bool A_IS_F32, bool OUT_IS_F32>
__global__ __launch_bounds__(256)
void gemm_bf16_wmma(const void* __restrict__ Aptr,
                    const float* __restrict__ B,
                    void* __restrict__ Cptr,
                    const float* __restrict__ bias,
                    __bf16* __restrict__ vt,
                    int M, int N, int K) {
    constexpr int BM = 128, BN = 128, BK = 64;
    __shared__ __bf16 As [BM][BK + 8];   // [row][k]   128 x 72
    __shared__ __bf16 BsT[BN][BK + 8];   // [col][k]   128 x 72 (transposed)

    const int tid  = threadIdx.x;
    const int lane = tid & 31;
    const int wave = tid >> 5;
    const int wy   = wave >> 1;          // 0..3
    const int wx   = wave & 1;           // 0..1
    const int brow = blockIdx.y * BM;
    const int bcol = blockIdx.x * BN;
    const int m    = lane & 15;
    const int half = lane >> 4;

    v8f zero = {};
    v8f acc[2][4];
    #pragma unroll
    for (int i = 0; i < 2; ++i)
        #pragma unroll
        for (int j = 0; j < 4; ++j) acc[i][j] = zero;

    // staging assignments
    const int ar  = tid >> 1;            // A: 2 threads per row, 32 elems each
    const int ac0 = (tid & 1) * 32;
    const int bk2 = (tid >> 3) * 2;      // B: a pair of k-rows ...
    const int bc0 = (tid & 7) * 16;      // ... x 16 cols per thread

    // register staging for the software pipeline
    f32x4 aregf[8];                      // A tile slice as f32 (A_IS_F32)
    u32x4 aregh[4];                      // A tile slice as bf16 (else)
    f32x4 breg0[4], breg1[4];            // two B k-rows, 16 f32 each

    auto load_tile = [&](int kk) {
        if constexpr (A_IS_F32) {
            const float* Ar = (const float*)Aptr + (size_t)(brow + ar) * K + kk + ac0;
            #pragma unroll
            for (int i = 0; i < 8; ++i) aregf[i] = *(const f32x4*)(Ar + 4 * i);
        } else {
            const __bf16* Ar = (const __bf16*)Aptr + (size_t)(brow + ar) * K + kk + ac0;
            #pragma unroll
            for (int i = 0; i < 4; ++i) aregh[i] = *(const u32x4*)(Ar + 8 * i);
        }
        const float* Br0 = B + (size_t)(kk + bk2) * N + bcol + bc0;
        const float* Br1 = Br0 + N;
        #pragma unroll
        for (int i = 0; i < 4; ++i) {
            breg0[i] = *(const f32x4*)(Br0 + 4 * i);
            breg1[i] = *(const f32x4*)(Br1 + 4 * i);
        }
    };

    load_tile(0);

    for (int k0 = 0; k0 < K; k0 += BK) {
        // drain staged registers into LDS
        if constexpr (A_IS_F32) {
            #pragma unroll
            for (int i = 0; i < 8; ++i) {
                *(unsigned int*)&As[ar][ac0 + 4 * i]     = pack2(aregf[i].x, aregf[i].y);
                *(unsigned int*)&As[ar][ac0 + 4 * i + 2] = pack2(aregf[i].z, aregf[i].w);
            }
        } else {
            #pragma unroll
            for (int i = 0; i < 4; ++i)
                *(u32x4*)&As[ar][ac0 + 8 * i] = aregh[i];
        }
        #pragma unroll
        for (int i = 0; i < 4; ++i) {
            *(unsigned int*)&BsT[bc0 + 4 * i + 0][bk2] = pack2(breg0[i].x, breg1[i].x);
            *(unsigned int*)&BsT[bc0 + 4 * i + 1][bk2] = pack2(breg0[i].y, breg1[i].y);
            *(unsigned int*)&BsT[bc0 + 4 * i + 2][bk2] = pack2(breg0[i].z, breg1[i].z);
            *(unsigned int*)&BsT[bc0 + 4 * i + 3][bk2] = pack2(breg0[i].w, breg1[i].w);
        }
        __syncthreads();

        // issue next tile's global loads (overlap with WMMA below)
        const int kn = (k0 + BK < K) ? (k0 + BK) : k0;
        load_tile(kn);

        #pragma unroll
        for (int ks = 0; ks < BK; ks += 32) {
            // A fragments (16x32): 16 contiguous k at ks+half*16
            Frag16 afr[2];
            #pragma unroll
            for (int mt = 0; mt < 2; ++mt) {
                const int rowb = wy * 32 + mt * 16 + m;
                afr[mt].q[0] = *(const u32x4*)&As[rowb][ks + half * 16];
                afr[mt].q[1] = *(const u32x4*)&As[rowb][ks + half * 16 + 8];
            }
            // B fragments (32x16): 16 contiguous k at ks+half*16 in BsT rows
            #pragma unroll
            for (int nt = 0; nt < 4; ++nt) {
                Frag16 bfr;
                const int colb = wx * 64 + nt * 16 + m;
                bfr.q[0] = *(const u32x4*)&BsT[colb][ks + half * 16];
                bfr.q[1] = *(const u32x4*)&BsT[colb][ks + half * 16 + 8];
                #pragma unroll
                for (int mt = 0; mt < 2; ++mt)
                    acc[mt][nt] = wmma_bf16(afr[mt].v, bfr.v, acc[mt][nt]);
            }
        }
        __syncthreads();
    }

    // C/D layout: VGPR j holds row M = half*8 + j, col = lane&15
    #pragma unroll
    for (int nt = 0; nt < 4; ++nt) {
        const int col = bcol + wx * 64 + nt * 16 + m;
        if constexpr (OUT_IS_F32) {
            const float bv = bias[col];
            #pragma unroll
            for (int mt = 0; mt < 2; ++mt)
                #pragma unroll
                for (int j = 0; j < 8; ++j) {
                    const int row = brow + wy * 32 + mt * 16 + half * 8 + j;
                    ((float*)Cptr)[(size_t)row * N + col] = acc[mt][nt][j] + bv;
                }
        } else {
            // wave-uniform predicate: 2048 is a multiple of 16, m < 16
            const bool use_vt = (vt != nullptr) && (col >= 2048);
            if (use_vt) {
                const int dcol = col - 2048;
                #pragma unroll
                for (int mt = 0; mt < 2; ++mt)
                    #pragma unroll
                    for (int j = 0; j < 8; ++j) {
                        const int row = brow + wy * 32 + mt * 16 + half * 8 + j;
                        const int bb = row >> 12, ss = row & 4095;
                        vt[((size_t)bb * 1024 + dcol) * 4096 + ss] = (__bf16)acc[mt][nt][j];
                    }
            } else {
                #pragma unroll
                for (int mt = 0; mt < 2; ++mt)
                    #pragma unroll
                    for (int j = 0; j < 8; ++j) {
                        const int row = brow + wy * 32 + mt * 16 + half * 8 + j;
                        ((__bf16*)Cptr)[(size_t)row * N + col] = (__bf16)acc[mt][nt][j];
                    }
            }
        }
    }
}

// ---------------------------------------------------------------------------
// Flash attention over full D=1024 (reference semantics, scale = 1/sqrt(128)).
// KQV layout: [B*S, 3072] bf16 with K at col 0, Q at 1024 (V third unused);
// Vt layout: [4][1024][4096] bf16 (keys contiguous -> pair loads).
// One block = 16 query rows; 8 waves each own a disjoint 128-wide D slice.
// Partial scores are reduced across waves with LDS float atomics.
// All K/V fragments of a k-tile are loaded before their WMMAs so the loads
// issue as one clause and drain incrementally instead of wait-0 per WMMA.
// ---------------------------------------------------------------------------
__global__ __launch_bounds__(256)
void flash_attn(const __bf16* __restrict__ KQV, const __bf16* __restrict__ Vt,
                __bf16* __restrict__ O) {
    constexpr int S = 4096, D3 = 3072, D = 1024;
    const int blk   = blockIdx.x;         // 1024 blocks
    const int b     = blk >> 8;           // 256 q-tiles per batch
    const int qbase = (blk & 255) * 16;
    const int tid   = threadIdx.x;
    const int lane  = tid & 31;
    const int wave  = tid >> 5;
    const int dbase = wave * 128;
    const int m     = lane & 15;
    const int half  = lane >> 4;

    __shared__ float  S_lds[16][32];
    __shared__ __bf16 P_lds[16][32];
    __shared__ float  mstat[16], lstat[16], alpha_lds[16];

    const float scale = 0.088388347648318447f;   // 1/sqrt(1024/8)

    // Preload this wave's Q slice (16 rows x 128 cols) as 4 A-fragments.
    Frag16 qf[4];
    const __bf16* Qrow = KQV + (size_t)(b * S + qbase + m) * D3 + D + dbase;
    #pragma unroll
    for (int f = 0; f < 4; ++f) {
        qf[f].q[0] = *(const u32x4*)(Qrow + f * 32 + half * 16);
        qf[f].q[1] = *(const u32x4*)(Qrow + f * 32 + half * 16 + 8);
    }

    v8f zero = {};
    v8f o_acc[8];
    #pragma unroll
    for (int t = 0; t < 8; ++t) o_acc[t] = zero;

    if (tid < 16) { mstat[tid] = -1e30f; lstat[tid] = 0.0f; }
    __syncthreads();

    for (int kb = 0; kb < S; kb += 32) {
        // zero partial-score tile
        ((float*)S_lds)[tid]       = 0.0f;
        ((float*)S_lds)[tid + 256] = 0.0f;
        __syncthreads();

        // ---- load ALL K fragments of this k-tile, then run the 8 WMMAs ----
        Frag16 kf[2][4];
        #pragma unroll
        for (int nt = 0; nt < 2; ++nt) {
            const __bf16* Krow = KQV + (size_t)(b * S + kb + nt * 16 + m) * D3 + dbase;
            #pragma unroll
            for (int f = 0; f < 4; ++f) {
                kf[nt][f].q[0] = *(const u32x4*)(Krow + f * 32 + half * 16);
                kf[nt][f].q[1] = *(const u32x4*)(Krow + f * 32 + half * 16 + 8);
            }
        }
        // unconditional speculative prefetch of next k-tile (dropped if invalid)
        __builtin_prefetch(KQV + (size_t)(b * S + kb + 32 + m) * D3 + dbase, 0, 1);

        v8f sacc[2]; sacc[0] = zero; sacc[1] = zero;
        #pragma unroll
        for (int nt = 0; nt < 2; ++nt)
            #pragma unroll
            for (int f = 0; f < 4; ++f)
                sacc[nt] = wmma_bf16(qf[f].v, kf[nt][f].v, sacc[nt]);

        // ---- issue V fragment loads now: latency hides under softmax ----
        Frag16 vf[8];
        #pragma unroll
        for (int nt = 0; nt < 8; ++nt) {
            const int dcol = dbase + nt * 16 + m;
            const __bf16* Vtrow = Vt + ((size_t)b * 1024 + dcol) * 4096 + kb;
            vf[nt].q[0] = *(const u32x4*)(Vtrow + half * 16);
            vf[nt].q[1] = *(const u32x4*)(Vtrow + half * 16 + 8);
        }

        // cross-wave reduction of partial scores
        #pragma unroll
        for (int nt = 0; nt < 2; ++nt)
            #pragma unroll
            for (int j = 0; j < 8; ++j)
                atomicAdd(&S_lds[half * 8 + j][nt * 16 + m], sacc[nt][j]);
        __syncthreads();

        // online softmax update (one thread per q-row)
        if (tid < 16) {
            const float mo = mstat[tid];
            float mx = mo;
            #pragma unroll
            for (int k = 0; k < 32; ++k)
                mx = fmaxf(mx, S_lds[tid][k] * scale);
            const float al = __expf(mo - mx);
            float ls = 0.0f;
            #pragma unroll
            for (int k = 0; k < 32; ++k) {
                const float p = __expf(S_lds[tid][k] * scale - mx);
                P_lds[tid][k] = (__bf16)p;
                ls += p;
            }
            lstat[tid]     = lstat[tid] * al + ls;
            mstat[tid]     = mx;
            alpha_lds[tid] = al;
        }
        __syncthreads();

        // rescale O accumulators (per-row alpha; row of VGPR j = half*8+j)
        float alj[8];
        #pragma unroll
        for (int j = 0; j < 8; ++j) alj[j] = alpha_lds[half * 8 + j];
        #pragma unroll
        for (int nt = 0; nt < 8; ++nt)
            #pragma unroll
            for (int j = 0; j < 8; ++j) o_acc[nt][j] *= alj[j];

        // P fragment (16q x 32keys), A layout: 16 contiguous keys at half*16
        Frag16 pf;
        pf.q[0] = *(const u32x4*)&P_lds[m][half * 16];
        pf.q[1] = *(const u32x4*)&P_lds[m][half * 16 + 8];

        // O += P @ V over this wave's 128-wide D slice (8 n-tiles)
        #pragma unroll
        for (int nt = 0; nt < 8; ++nt)
            o_acc[nt] = wmma_bf16(pf.v, vf[nt].v, o_acc[nt]);
    }

    // finalize: divide by row sums, emit bf16
    float linv[8];
    #pragma unroll
    for (int j = 0; j < 8; ++j) linv[j] = 1.0f / lstat[half * 8 + j];
    #pragma unroll
    for (int nt = 0; nt < 8; ++nt)
        #pragma unroll
        for (int j = 0; j < 8; ++j) {
            const int row = b * S + qbase + half * 8 + j;
            const int col = dbase + nt * 16 + m;
            O[(size_t)row * D + col] = (__bf16)(o_acc[nt][j] * linv[j]);
        }
}

// ---------------------------------------------------------------------------
// Launch: GEMM1 (X @ W_qkv -> K,Q rows + V transposed), flash attention,
// GEMM3 (+bias, f32 out).
// ---------------------------------------------------------------------------
extern "C" void kernel_launch(void* const* d_in, const int* in_sizes, int n_in,
                              void* d_out, int out_size, void* d_ws, size_t ws_size,
                              hipStream_t stream) {
    (void)in_sizes; (void)n_in; (void)out_size; (void)ws_size;
    const float* x    = (const float*)d_in[0];  // [4,4096,2048]
    const float* wqkv = (const float*)d_in[1];  // [2048,3072]
    const float* wff  = (const float*)d_in[2];  // [1024,1024]
    const float* bff  = (const float*)d_in[3];  // [1024]
    float* out = (float*)d_out;                 // [4,4096,1024] f32

    const int M = 4 * 4096;                     // 16384 rows
    __bf16* kqv  = (__bf16*)d_ws;               // [16384,3072] bf16 (96 MB; V third unused)
    __bf16* vtp  = kqv + (size_t)M * 3072;      // [4][1024][4096] bf16 (32 MB)
    __bf16* o_bf = vtp + (size_t)4 * 1024 * 4096; // [16384,1024] bf16 (32 MB)

    dim3 blk(256);
    gemm_bf16_wmma<true, false><<<dim3(3072 / 128, M / 128), blk, 0, stream>>>(
        x, wqkv, kqv, nullptr, vtp, M, 3072, 2048);
    flash_attn<<<dim3(M / 16), blk, 0, stream>>>(kqv, vtp, o_bf);
    gemm_bf16_wmma<false, true><<<dim3(1024 / 128, M / 128), blk, 0, stream>>>(
        o_bf, wff, out, bff, nullptr, M, 1024, 1024);
}
